// RNN_Model_41644002902418
// MI455X (gfx1250) — matrix operand order repaired and checked by reading, not compiled, not run on hardware
//
#include <hip/hip_runtime.h>
#include <hip/hip_bf16.h>

typedef __attribute__((ext_vector_type(16))) _Float16 v16h;
typedef __attribute__((ext_vector_type(8)))  _Float16 v8h;
typedef __attribute__((ext_vector_type(8)))  float    v8f;

#define TT   300
#define DIN  39
#define H1   35
#define H2   30
#define BB   4096
#define FRAG 512   // halves per 16x16x32 B-fragment (32 lanes x 16 halves)

// fragment counts: K padded to multiples of 32, N padded to multiples of 16
#define NF_W1  (3*9)   // K 74->96, N 140->144
#define NF_W2  (3*8)   // K 65->96, N 120->128
#define NF_FC1 (2*4)   // K 60->64, N 50->64
#define NF_FC2 (2*3)   // K 50->64, N 40->48
#define NF_OUT (2*1)   // K 40->64, N 2->16

#define OFF_FW1 0
#define OFF_FW2 (OFF_FW1 + NF_W1*FRAG)
#define OFF_BW1 (OFF_FW2 + NF_W2*FRAG)
#define OFF_BW2 (OFF_BW1 + NF_W1*FRAG)
#define OFF_FC1 (OFF_BW2 + NF_W2*FRAG)
#define OFF_FC2 (OFF_FC1 + NF_FC1*FRAG)
#define OFF_OUT (OFF_FC2 + NF_FC2*FRAG)
#define HBUF_BYTE_OFF 131072  // f16 hidden buffer starts here in d_ws

__device__ __forceinline__ float sigm(float v) { return 1.f / (1.f + __expf(-v)); }

// ---------------------------------------------------------------------------
// Swizzle f32 row-major W[K][N] into f16 WMMA B-fragments:
//   frag (kc,nt): lane l holds column n = nt*16 + (l&15),
//   halves e=0..15 hold k = kc*32 + ((e>>3)<<4) + ((l>>4)<<3) + (e&7). Zero-padded OOB.
// ---------------------------------------------------------------------------
__device__ void swz(const float* __restrict__ W, int K, int N, int KC, int NT,
                    _Float16* __restrict__ dst, int tid, int stride) {
  int total = KC * NT * FRAG;
  for (int i = tid; i < total; i += stride) {
    int frag = i >> 9, r = i & 511;
    int lane = r >> 4, e = r & 15;
    int kc = frag / NT, nt = frag - kc * NT;
    int k = kc * 32 + ((e >> 3) << 4) + ((lane >> 4) << 3) + (e & 7);
    int n = nt * 16 + (lane & 15);
    float v = (k < K && n < N) ? W[k * N + n] : 0.f;
    dst[i] = (_Float16)v;
  }
}

__global__ void k_swizzle(const float* fw1, const float* fw2,
                          const float* bw1, const float* bw2,
                          const float* wfc1, const float* wfc2,
                          const float* wout, _Float16* dst) {
  int tid = blockIdx.x * blockDim.x + threadIdx.x;
  int stride = gridDim.x * blockDim.x;
  swz(fw1, DIN + H1, 4 * H1, 3, 9, dst + OFF_FW1, tid, stride);
  swz(fw2, H1 + H2, 4 * H2, 3, 8, dst + OFF_FW2, tid, stride);
  swz(bw1, DIN + H1, 4 * H1, 3, 9, dst + OFF_BW1, tid, stride);
  swz(bw2, H1 + H2, 4 * H2, 3, 8, dst + OFF_BW2, tid, stride);
  swz(wfc1, 60, 50, 2, 4, dst + OFF_FC1, tid, stride);
  swz(wfc2, 50, 40, 2, 3, dst + OFF_FC2, tid, stride);
  swz(wout, 40, 2, 2, 1, dst + OFF_OUT, tid, stride);
}

// ---------------------------------------------------------------------------
// Fused 2-layer LSTM over T steps. One wave = 16 batch rows, full recurrence.
// blockDim = 64 (2 waves). gridDim = (BB/32, 2 directions).
// Weights live in LDS as pre-swizzled B-fragments; h/c state stays on-chip.
// ---------------------------------------------------------------------------
__global__ __launch_bounds__(64) void k_lstm(
    const float* __restrict__ x,
    const float* __restrict__ fb1, const float* __restrict__ fb2,
    const float* __restrict__ bb1, const float* __restrict__ bb2,
    const _Float16* __restrict__ wfrag,
    _Float16* __restrict__ hbuf) {
  __shared__ alignas(32) _Float16 sW1[NF_W1 * FRAG];   // 27.0 KB
  __shared__ alignas(32) _Float16 sW2[NF_W2 * FRAG];   // 24.0 KB
  __shared__ alignas(32) _Float16 sA1[2][16 * 96];     // [x(39) | h1(35) | pad] f16
  __shared__ alignas(32) _Float16 sA2[2][16 * 96];     // [h1(35) | h2(30) | pad] f16
  __shared__ float sZ[2][16 * 144];                    // gate pre-activations
  __shared__ float sC1[2][16 * 36];
  __shared__ float sC2[2][16 * 32];

  const int tid = threadIdx.x;
  const int wave = tid >> 5;
  const int lane = tid & 31;
  const int dir = blockIdx.y;
  const int b0 = (blockIdx.x * 2 + wave) * 16;
  const int mrow = lane & 15;
  const int hi = lane >> 4;

  const _Float16* gW1 = wfrag + (dir ? OFF_BW1 : OFF_FW1);
  const _Float16* gW2 = wfrag + (dir ? OFF_BW2 : OFF_FW2);
  for (int i = tid; i < NF_W1 * FRAG; i += 64) sW1[i] = gW1[i];
  for (int i = tid; i < NF_W2 * FRAG; i += 64) sW2[i] = gW2[i];
  for (int i = lane; i < 16 * 96; i += 32) {
    sA1[wave][i] = (_Float16)0.f;
    sA2[wave][i] = (_Float16)0.f;
  }
  for (int i = lane; i < 16 * 36; i += 32) sC1[wave][i] = 0.f;
  for (int i = lane; i < 16 * 32; i += 32) sC2[wave][i] = 0.f;
  __syncthreads();

  const float* b1 = dir ? bb1 : fb1;
  const float* b2 = dir ? bb2 : fb2;
  _Float16* hout = hbuf + (size_t)dir * (size_t)BB * TT * 32;

  for (int t = 0; t < TT; ++t) {
    const int ts = dir ? (TT - 1 - t) : t;
    // stage x tile (16 rows x 39) as f16 into A1 columns 0..38
    for (int i = lane; i < 16 * DIN; i += 32) {
      int m = i / DIN, d = i - m * DIN;
      sA1[wave][m * 96 + d] = (_Float16)x[((size_t)(b0 + m) * TT + ts) * DIN + d];
    }
    if (t + 1 < TT) {
      int tn = dir ? (TT - 2 - t) : (t + 1);
      __builtin_prefetch(&x[((size_t)(b0 + mrow) * TT + tn) * DIN], 0, 0);
    }
    __syncthreads();

    // ---- layer 1: z[16x140] = A1[16x96(pad)] x W1 ----
    v8f acc[9] = {};
#pragma unroll
    for (int kc = 0; kc < 3; ++kc) {
      const _Float16* ap = &sA1[wave][mrow * 96 + kc * 32 + (hi << 3)];
      v8h lo = *(const v8h*)ap;
      v8h up = *(const v8h*)(ap + 16);
      v16h a;
#pragma unroll
      for (int e = 0; e < 8; ++e) { a[e] = lo[e]; a[e + 8] = up[e]; }
#pragma unroll
      for (int n = 0; n < 9; ++n) {
        v16h bf = *(const v16h*)&sW1[(kc * 9 + n) * FRAG + lane * 16];
        acc[n] = __builtin_amdgcn_wmma_f32_16x16x32_f16(
            false, a, false, bf, (short)0, acc[n], false, false);
      }
    }
#pragma unroll
    for (int n = 0; n < 9; ++n)
#pragma unroll
      for (int r = 0; r < 8; ++r)
        sZ[wave][(r + hi * 8) * 144 + n * 16 + mrow] = acc[n][r];
    __syncthreads();

    // ---- layer-1 gates (i, j, f, o) ----
    for (int i = lane; i < 16 * H1; i += 32) {
      int m = i / H1, u = i - m * H1;
      const float* zr = &sZ[wave][m * 144];
      float zi = zr[u]          + b1[u];
      float zj = zr[H1 + u]     + b1[H1 + u];
      float zf = zr[2 * H1 + u] + b1[2 * H1 + u];
      float zo = zr[3 * H1 + u] + b1[3 * H1 + u];
      float c = sC1[wave][m * 36 + u];
      c = sigm(zf + 1.f) * c + sigm(zi) * tanhf(zj);
      float h = sigm(zo) * tanhf(c);
      sC1[wave][m * 36 + u] = c;
      _Float16 hh = (_Float16)h;
      sA1[wave][m * 96 + DIN + u] = hh;  // feed next step's layer-1 input
      sA2[wave][m * 96 + u] = hh;        // feed layer 2
    }
    __syncthreads();

    // ---- layer 2: z[16x120] = A2[16x96(pad)] x W2 ----
    v8f acc2[8] = {};
#pragma unroll
    for (int kc = 0; kc < 3; ++kc) {
      const _Float16* ap = &sA2[wave][mrow * 96 + kc * 32 + (hi << 3)];
      v8h lo = *(const v8h*)ap;
      v8h up = *(const v8h*)(ap + 16);
      v16h a;
#pragma unroll
      for (int e = 0; e < 8; ++e) { a[e] = lo[e]; a[e + 8] = up[e]; }
#pragma unroll
      for (int n = 0; n < 8; ++n) {
        v16h bf = *(const v16h*)&sW2[(kc * 8 + n) * FRAG + lane * 16];
        acc2[n] = __builtin_amdgcn_wmma_f32_16x16x32_f16(
            false, a, false, bf, (short)0, acc2[n], false, false);
      }
    }
#pragma unroll
    for (int n = 0; n < 8; ++n)
#pragma unroll
      for (int r = 0; r < 8; ++r)
        sZ[wave][(r + hi * 8) * 144 + n * 16 + mrow] = acc2[n][r];
    __syncthreads();

    // ---- layer-2 gates; emit h2 (f16) ----
    for (int i = lane; i < 16 * H2; i += 32) {
      int m = i / H2, u = i - m * H2;
      const float* zr = &sZ[wave][m * 144];
      float zi = zr[u]          + b2[u];
      float zj = zr[H2 + u]     + b2[H2 + u];
      float zf = zr[2 * H2 + u] + b2[2 * H2 + u];
      float zo = zr[3 * H2 + u] + b2[3 * H2 + u];
      float c = sC2[wave][m * 32 + u];
      c = sigm(zf + 1.f) * c + sigm(zi) * tanhf(zj);
      float h = sigm(zo) * tanhf(c);
      sC2[wave][m * 32 + u] = c;
      _Float16 hh = (_Float16)h;
      sA2[wave][m * 96 + H1 + u] = hh;  // feed next step's layer-2 state
      hout[((size_t)(b0 + m) * TT + ts) * 32 + u] = hh;
    }
    __syncthreads();
  }
}

// ---------------------------------------------------------------------------
// Fused FC stack: [B*T,60] -> relu50 -> relu40 -> 2. One wave = 16 rows.
// blockDim = 128 (4 waves). gridDim = B*T/64.
// ---------------------------------------------------------------------------
__global__ __launch_bounds__(128) void k_fc(
    const _Float16* __restrict__ hbuf,
    const _Float16* __restrict__ wfrag,
    const float* __restrict__ bfc1, const float* __restrict__ bfc2,
    const float* __restrict__ bout,
    float* __restrict__ out) {
  __shared__ alignas(32) _Float16 sWf[(NF_FC1 + NF_FC2 + NF_OUT) * FRAG];
  __shared__ alignas(32) _Float16 sA[4][16 * 64];
  __shared__ float sZ[4][16 * 64];

  const int tid = threadIdx.x;
  const int wave = tid >> 5;
  const int lane = tid & 31;
  const int mrow = lane & 15;
  const int hi = lane >> 4;

  for (int i = tid; i < (NF_FC1 + NF_FC2 + NF_OUT) * FRAG; i += 128)
    sWf[i] = wfrag[OFF_FC1 + i];
  const _Float16* sFC1 = sWf;
  const _Float16* sFC2 = sWf + NF_FC1 * FRAG;
  const _Float16* sOut = sWf + (NF_FC1 + NF_FC2) * FRAG;

  const size_t r0 = ((size_t)blockIdx.x * 4 + wave) * 16;
  const _Float16* hf = hbuf;
  const _Float16* hb = hbuf + (size_t)BB * TT * 32;

  // stage input rows: concat [h_fw(30) | h_bw(30) | pad]
  for (int i = lane; i < 16 * 64; i += 32) {
    int m = i >> 6, u = i & 63;
    _Float16 v = (_Float16)0.f;
    if (u < 30)      v = hf[(r0 + m) * 32 + u];
    else if (u < 60) v = hb[(r0 + m) * 32 + (u - 30)];
    sA[wave][i] = v;
  }
  __syncthreads();

  // ---- fc1: 60 -> 50, relu ----
  v8f a1[4] = {};
#pragma unroll
  for (int kc = 0; kc < 2; ++kc) {
    const _Float16* ap = &sA[wave][mrow * 64 + kc * 32 + (hi << 3)];
    v8h lo = *(const v8h*)ap;
    v8h up = *(const v8h*)(ap + 16);
    v16h a;
#pragma unroll
    for (int e = 0; e < 8; ++e) { a[e] = lo[e]; a[e + 8] = up[e]; }
#pragma unroll
    for (int n = 0; n < 4; ++n) {
      v16h bf = *(const v16h*)&sFC1[(kc * 4 + n) * FRAG + lane * 16];
      a1[n] = __builtin_amdgcn_wmma_f32_16x16x32_f16(
          false, a, false, bf, (short)0, a1[n], false, false);
    }
  }
#pragma unroll
  for (int n = 0; n < 4; ++n)
#pragma unroll
    for (int r = 0; r < 8; ++r)
      sZ[wave][(r + hi * 8) * 64 + n * 16 + mrow] = a1[n][r];
  __syncthreads();
  for (int i = lane; i < 16 * 64; i += 32) {
    int m = i >> 6, u = i & 63;
    float v = 0.f;
    if (u < 50) { v = sZ[wave][m * 64 + u] + bfc1[u]; v = v > 0.f ? v : 0.f; }
    sA[wave][i] = (_Float16)v;
  }
  __syncthreads();

  // ---- fc2: 50 -> 40, relu ----
  v8f a2[3] = {};
#pragma unroll
  for (int kc = 0; kc < 2; ++kc) {
    const _Float16* ap = &sA[wave][mrow * 64 + kc * 32 + (hi << 3)];
    v8h lo = *(const v8h*)ap;
    v8h up = *(const v8h*)(ap + 16);
    v16h a;
#pragma unroll
    for (int e = 0; e < 8; ++e) { a[e] = lo[e]; a[e + 8] = up[e]; }
#pragma unroll
    for (int n = 0; n < 3; ++n) {
      v16h bf = *(const v16h*)&sFC2[(kc * 3 + n) * FRAG + lane * 16];
      a2[n] = __builtin_amdgcn_wmma_f32_16x16x32_f16(
          false, a, false, bf, (short)0, a2[n], false, false);
    }
  }
#pragma unroll
  for (int n = 0; n < 3; ++n)
#pragma unroll
    for (int r = 0; r < 8; ++r)
      sZ[wave][(r + hi * 8) * 64 + n * 16 + mrow] = a2[n][r];
  __syncthreads();
  for (int i = lane; i < 16 * 64; i += 32) {
    int m = i >> 6, u = i & 63;
    float v = 0.f;
    if (u < 40) { v = sZ[wave][m * 64 + u] + bfc2[u]; v = v > 0.f ? v : 0.f; }
    sA[wave][i] = (_Float16)v;
  }
  __syncthreads();

  // ---- out: 40 -> 2 ----
  v8f a3 = {};
#pragma unroll
  for (int kc = 0; kc < 2; ++kc) {
    const _Float16* ap = &sA[wave][mrow * 64 + kc * 32 + (hi << 3)];
    v8h lo = *(const v8h*)ap;
    v8h up = *(const v8h*)(ap + 16);
    v16h a;
#pragma unroll
    for (int e = 0; e < 8; ++e) { a[e] = lo[e]; a[e + 8] = up[e]; }
    v16h bf = *(const v16h*)&sOut[kc * FRAG + lane * 16];
    a3 = __builtin_amdgcn_wmma_f32_16x16x32_f16(
        false, a, false, bf, (short)0, a3, false, false);
  }
#pragma unroll
  for (int r = 0; r < 8; ++r)
    sZ[wave][(r + hi * 8) * 64 + mrow] = a3[r];
  __syncthreads();
  {
    int m = lane >> 1, o = lane & 1;  // 32 lanes cover 16 rows x 2 outputs
    out[(r0 + m) * 2 + o] = sZ[wave][m * 64 + o] + bout[o];
  }
}

// ---------------------------------------------------------------------------
extern "C" void kernel_launch(void* const* d_in, const int* in_sizes, int n_in,
                              void* d_out, int out_size, void* d_ws, size_t ws_size,
                              hipStream_t stream) {
  (void)in_sizes; (void)n_in; (void)out_size; (void)ws_size;
  const float* x     = (const float*)d_in[0];
  const float* fw_W1 = (const float*)d_in[1];
  const float* fw_b1 = (const float*)d_in[2];
  const float* fw_W2 = (const float*)d_in[3];
  const float* fw_b2 = (const float*)d_in[4];
  const float* bw_W1 = (const float*)d_in[5];
  const float* bw_b1 = (const float*)d_in[6];
  const float* bw_W2 = (const float*)d_in[7];
  const float* bw_b2 = (const float*)d_in[8];
  const float* W_fc1 = (const float*)d_in[9];
  const float* b_fc1 = (const float*)d_in[10];
  const float* W_fc2 = (const float*)d_in[11];
  const float* b_fc2 = (const float*)d_in[12];
  const float* W_out = (const float*)d_in[13];
  const float* b_out = (const float*)d_in[14];

  _Float16* wfrag = (_Float16*)d_ws;
  _Float16* hbuf  = (_Float16*)((char*)d_ws + HBUF_BYTE_OFF);

  k_swizzle<<<64, 256, 0, stream>>>(fw_W1, fw_W2, bw_W1, bw_W2,
                                    W_fc1, W_fc2, W_out, wfrag);
  k_lstm<<<dim3(BB / 32, 2), 64, 0, stream>>>(x, fw_b1, fw_b2, bw_b1, bw_b2,
                                              wfrag, hbuf);
  k_fc<<<dim3((BB * TT) / 64), 128, 0, stream>>>(hbuf, wfrag, b_fc1, b_fc2,
                                                 b_out, (float*)d_out);
}
